// SelfAttention_34892314313227
// MI455X (gfx1250) — matrix operand order repaired and checked by reading, not compiled, not run on hardware
//
#include <hip/hip_runtime.h>
#include <hip/hip_bf16.h>

// Problem constants (from reference): B=2, S=2048, E=1024, H=16, D=64
#define BB 2
#define SS 2048
#define EE 1024
#define HH 16
#define DD 64
#define BS (BB * SS)

typedef __bf16 bf16_t;
typedef __attribute__((ext_vector_type(16))) __bf16 v16bf;
typedef __attribute__((ext_vector_type(8)))  __bf16 v8bf;
typedef __attribute__((ext_vector_type(4)))  __bf16 v4bf;
typedef __attribute__((ext_vector_type(8)))  float  v8f;

// ---------------------------------------------------------------------------
// WMMA operand layouts used here (contiguous k, not AMD's table order; valid as A and B share it):
// A (16x32, 16-bit): lane g*16+ln holds row m=ln; elems i = K base+i,
//   base = g*16 (contiguous, matching B).
// B (32x16, 16-bit): lane holds column n=ln; elems i = K g*16+i (contiguous).
// ---------------------------------------------------------------------------
static __device__ __forceinline__ v16bf load_a_bf(const bf16_t* base) {
  v8bf lo = *(const v8bf*)(base);        // K chunk [0..7]
  v8bf hi = *(const v8bf*)(base + 8);    // K chunk [8..15]
  v16bf r;
#pragma unroll
  for (int i = 0; i < 8; ++i) { r[i] = lo[i]; r[i + 8] = hi[i]; }
  return r;
}

// ---------------------------------------------------------------------------
// Kernel 0: one-shot f32 -> bf16 conversion (bandwidth-bound pre-pass).
// ---------------------------------------------------------------------------
__global__ __launch_bounds__(256) void cvt_bf16_kernel(
    const float* __restrict__ in, bf16_t* __restrict__ out, int n4) {
  const int i = blockIdx.x * 256 + threadIdx.x;
  if (i < n4) {
    const float4 x = ((const float4*)in)[i];
    v4bf y;
    y[0] = (__bf16)x.x; y[1] = (__bf16)x.y;
    y[2] = (__bf16)x.z; y[3] = (__bf16)x.w;
    ((v4bf*)out)[i] = y;
  }
}

// ---------------------------------------------------------------------------
// Kernel 1: projection GEMM  out = X @ W^T + b  (torch Linear), bf16 inputs.
// C[m][n] = sum_k X[m][k] * W[n][k] -- both operands are K-contiguous rows,
// matching the A/B WMMA layouts directly. Each wave computes a 32x64 tile:
// per k-step, 2 A-operands x 4 B-operands feed 8 WMMAs from 12 b128 loads
// (1.5 loads per WMMA, 8 independent WMMAs to hide load latency behind).
// Output: transposed==0 -> [B,H,S,D] bf16 ; transposed==1 -> [B,H,D,S] bf16.
// ---------------------------------------------------------------------------
__global__ __launch_bounds__(128) void qkv_gemm_kernel(
    const bf16_t* __restrict__ Xb, const bf16_t* __restrict__ Wb,
    const float* __restrict__ bias, bf16_t* __restrict__ out, int transposed) {
  const int lane = threadIdx.x & 31;
  const int wave = threadIdx.x >> 5;
  const int g = lane >> 4, ln = lane & 15;
  const int m0 = blockIdx.x * 32;
  const int n0 = (blockIdx.y * 4 + wave) * 64;

  const bf16_t* xr0 = Xb + (size_t)(m0 + ln) * EE;
  const bf16_t* xr1 = Xb + (size_t)(m0 + 16 + ln) * EE;
  const bf16_t* wr[4];
#pragma unroll
  for (int nt = 0; nt < 4; ++nt) wr[nt] = Wb + (size_t)(n0 + nt * 16 + ln) * EE;

  v8f acc[2][4];
#pragma unroll
  for (int mh = 0; mh < 2; ++mh)
#pragma unroll
    for (int nt = 0; nt < 4; ++nt) acc[mh][nt] = (v8f){};

  for (int k0 = 0; k0 < EE; k0 += 32) {
    // Batch all operand loads first, then the 8 WMMAs.
    const v16bf a0 = load_a_bf(xr0 + k0 + g * 16);
    const v16bf a1 = load_a_bf(xr1 + k0 + g * 16);
    v16bf b[4];
#pragma unroll
    for (int nt = 0; nt < 4; ++nt) b[nt] = *(const v16bf*)(wr[nt] + k0 + g * 16);
#pragma unroll
    for (int nt = 0; nt < 4; ++nt) {
      acc[0][nt] = __builtin_amdgcn_wmma_f32_16x16x32_bf16(
          false, a0, false, b[nt], (short)0, acc[0][nt], false, false);
      acc[1][nt] = __builtin_amdgcn_wmma_f32_16x16x32_bf16(
          false, a1, false, b[nt], (short)0, acc[1][nt], false, false);
    }
  }

#pragma unroll
  for (int nt = 0; nt < 4; ++nt) {
    const int n = n0 + nt * 16 + ln;     // C/D layout: lane owns column n
    const float bn = bias[n];
    const int h = n >> 6, d = n & (DD - 1);
#pragma unroll
    for (int mh = 0; mh < 2; ++mh) {
#pragma unroll
      for (int v = 0; v < 8; ++v) {      // rows m = mh*16 + 8*g + v
        const int m = m0 + mh * 16 + g * 8 + v;
        const int b_ = m >> 11;          // m / S
        const int s = m & (SS - 1);      // m % S
        const float val = acc[mh][nt][v] + bn;
        const size_t idx = transposed
            ? ((size_t)(b_ * HH + h) * DD + d) * SS + s
            : ((size_t)(b_ * HH + h) * SS + s) * DD + d;
        out[idx] = (bf16_t)val;
      }
    }
  }
}

// ---------------------------------------------------------------------------
// Kernel 2: flash-style attention per (b,h), 64-key blocks.
// Q,K: [B,H,S,D] bf16 ; Vt: [B,H,D,S] bf16 ; out: [B,S,H,D] f32.
// Block = 128 threads = 4 waves; each wave owns one 16-row query tile.
// Per 64-key block: 8 score WMMAs + 8 PV WMMAs + 2 row-sum WMMAs per softmax
// update. Row sums are computed as P @ ones on the matrix unit: in the C/D
// layout each lane's 8 accumulator elements are exactly the per-row sums
// indexed by v -- replacing the sum butterfly (8 adds + 32 shuffles/blk).
// Softmax runs in the exp2 domain (1/sqrt(S)*log2 e folded into the scale ->
// exp2f = bare v_exp_f32). Max reduction: width-16 butterfly shuffles (a
// C-row spans 16 lanes). P transposed D-layout -> A-layout via wave-private
// LDS. Per-head K/V (512 KB) is L2-resident (192 MB L2).
// ---------------------------------------------------------------------------
__global__ __launch_bounds__(128) void attn_kernel(
    const bf16_t* __restrict__ Q, const bf16_t* __restrict__ K,
    const bf16_t* __restrict__ Vt, float* __restrict__ out) {
  __shared__ float psh[4][16][68];   // 64 cols + pad: 16B-aligned 8-col chunks

  const int lane = threadIdx.x & 31;
  const int wave = threadIdx.x >> 5;
  const int g = lane >> 4, ln = lane & 15;
  const int bh = blockIdx.y;
  const int b_ = bh >> 4, h = bh & (HH - 1);

  const bf16_t* Qh = Q + (size_t)bh * SS * DD;
  const bf16_t* Kh = K + (size_t)bh * SS * DD;
  const bf16_t* Vh = Vt + (size_t)bh * DD * SS;

  const int q0 = blockIdx.x * 64 + wave * 16;

  // Q tile (16 x 64) resident in registers, A-layout, K(=d) chunks of 32.
  v16bf qa[2];
#pragma unroll
  for (int c = 0; c < 2; ++c)
    qa[c] = load_a_bf(Qh + (size_t)(q0 + ln) * DD + c * 32 + g * 16);

  // Constant all-ones B operand for row-sum-via-WMMA.
  v16bf ones;
#pragma unroll
  for (int i = 0; i < 16; ++i) ones[i] = (__bf16)1.0f;

  // exp2 domain: scores scaled by (1/sqrt(2048)) * log2(e)
  const float ls = 0.022097086912079608f * 1.4426950408889634f;
  float mi[8], li[8];
  v8f oacc[4];
#pragma unroll
  for (int v = 0; v < 8; ++v) { mi[v] = -1e30f; li[v] = 0.0f; }
#pragma unroll
  for (int nt = 0; nt < 4; ++nt) oacc[nt] = (v8f){};

  float (*p)[68] = psh[wave];

  for (int kb = 0; kb < SS / 64; ++kb) {
    const int kbase = kb * 64;
    __builtin_prefetch(Kh + (size_t)(kbase + 64) * DD, 0, 1);
    __builtin_prefetch(Vh + (size_t)ln * SS + kbase + 64, 0, 1);

    // ---- scores: S = Q (16x64) @ K_block^T (64x64), four 16x16 n-tiles ----
    // Batch the 8 K-operand loads, then issue the 8 WMMAs back-to-back.
    v16bf kv[2][4];
#pragma unroll
    for (int c = 0; c < 2; ++c)
#pragma unroll
      for (int t = 0; t < 4; ++t)
        kv[c][t] =
            *(const v16bf*)(Kh + (size_t)(kbase + t * 16 + ln) * DD + c * 32 + g * 16);

    v8f sc[4];
#pragma unroll
    for (int t = 0; t < 4; ++t) sc[t] = (v8f){};
#pragma unroll
    for (int c = 0; c < 2; ++c)
#pragma unroll
      for (int t = 0; t < 4; ++t)
        sc[t] = __builtin_amdgcn_wmma_f32_16x16x32_bf16(
            false, qa[c], false, kv[c][t], (short)0, sc[t], false, false);

    // ---- online softmax over 64 columns (exp2 domain) ----
    float corr[8];
#pragma unroll
    for (int v = 0; v < 8; ++v) {
#pragma unroll
      for (int t = 0; t < 4; ++t) sc[t][v] *= ls;
      float mx = fmaxf(fmaxf(sc[0][v], sc[1][v]), fmaxf(sc[2][v], sc[3][v]));
#pragma unroll
      for (int o = 1; o < 16; o <<= 1) mx = fmaxf(mx, __shfl_xor(mx, o, 16));
      const float nm = fmaxf(mi[v], mx);
      corr[v] = exp2f(mi[v] - nm);
      mi[v] = nm;
    }
#pragma unroll
    for (int v = 0; v < 8; ++v)
#pragma unroll
      for (int t = 0; t < 4; ++t) sc[t][v] = exp2f(sc[t][v] - mi[v]);

#pragma unroll
    for (int nt = 0; nt < 4; ++nt)
#pragma unroll
      for (int v = 0; v < 8; ++v) oacc[nt][v] *= corr[v];

    // ---- transpose P (D-layout -> A-layout) through wave-private LDS ----
#pragma unroll
    for (int t = 0; t < 4; ++t)
#pragma unroll
      for (int v = 0; v < 8; ++v) p[8 * g + v][t * 16 + ln] = sc[t][v];
    __builtin_amdgcn_wave_barrier();     // same-wave DS ops are in-order

    v16bf pa[2];
#pragma unroll
    for (int c = 0; c < 2; ++c) {
      const float* pr = &p[ln][c * 32 + g * 16];
      const float4 c0 = *(const float4*)(pr);
      const float4 c1 = *(const float4*)(pr + 4);
      const float4 c2 = *(const float4*)(pr + 8);
      const float4 c3 = *(const float4*)(pr + 12);
      pa[c][0] = (__bf16)c0.x;  pa[c][1] = (__bf16)c0.y;
      pa[c][2] = (__bf16)c0.z;  pa[c][3] = (__bf16)c0.w;
      pa[c][4] = (__bf16)c1.x;  pa[c][5] = (__bf16)c1.y;
      pa[c][6] = (__bf16)c1.z;  pa[c][7] = (__bf16)c1.w;
      pa[c][8] = (__bf16)c2.x;  pa[c][9] = (__bf16)c2.y;
      pa[c][10] = (__bf16)c2.z; pa[c][11] = (__bf16)c2.w;
      pa[c][12] = (__bf16)c3.x; pa[c][13] = (__bf16)c3.y;
      pa[c][14] = (__bf16)c3.z; pa[c][15] = (__bf16)c3.w;
    }

    // ---- row sums on the matrix unit: lsum = P @ ones (64x16) ----
    // D-layout: every column identical; lane's elems [v] = row sums for
    // rows m = 8*g + v, exactly the indexing li[] uses.
    v8f lsum = (v8f){};
    lsum = __builtin_amdgcn_wmma_f32_16x16x32_bf16(
        false, pa[0], false, ones, (short)0, lsum, false, false);
    lsum = __builtin_amdgcn_wmma_f32_16x16x32_bf16(
        false, pa[1], false, ones, (short)0, lsum, false, false);
#pragma unroll
    for (int v = 0; v < 8; ++v) li[v] = li[v] * corr[v] + lsum[v];

    // ---- O += P (16x64) @ V_block (64x64): Vt rows are K-contiguous ----
    v16bf vv[2][4];
#pragma unroll
    for (int c = 0; c < 2; ++c)
#pragma unroll
      for (int nt = 0; nt < 4; ++nt)
        vv[c][nt] =
            *(const v16bf*)(Vh + (size_t)(nt * 16 + ln) * SS + kbase + c * 32 + g * 16);
#pragma unroll
    for (int c = 0; c < 2; ++c)
#pragma unroll
      for (int nt = 0; nt < 4; ++nt)
        oacc[nt] = __builtin_amdgcn_wmma_f32_16x16x32_bf16(
            false, pa[c], false, vv[c][nt], (short)0, oacc[nt], false, false);
  }

  // ---- epilogue: normalize by row sum, write [B,S,H,D] f32 ----
#pragma unroll
  for (int v = 0; v < 8; ++v) {
    const int s = q0 + 8 * g + v;
    const float inv = 1.0f / li[v];
#pragma unroll
    for (int nt = 0; nt < 4; ++nt) {
      const int d = nt * 16 + ln;
      out[(((size_t)b_ * SS + s) * HH + h) * DD + d] = oacc[nt][v] * inv;
    }
  }
}

// ---------------------------------------------------------------------------
// Launch: 6 conversion launches + 3 GEMMs + attention, all on `stream`.
// No allocs/syncs (graph-safe). Workspace (d_ws) layout, bf16 elements:
//   Xq|Xk|Xv: 3 * B*S*E   (3 * 8 MB)
//   Wq|Wk|Wv: 3 * E*E     (3 * 2 MB)
//   Q |K |Vt: 3 * B*H*S*D (3 * 8 MB)      total = 54 MB
// ---------------------------------------------------------------------------
extern "C" void kernel_launch(void* const* d_in, const int* in_sizes, int n_in,
                              void* d_out, int out_size, void* d_ws, size_t ws_size,
                              hipStream_t stream) {
  (void)in_sizes; (void)n_in; (void)out_size; (void)ws_size;
  const float* values  = (const float*)d_in[0];
  const float* keys    = (const float*)d_in[1];
  const float* queries = (const float*)d_in[2];
  const float* Wq = (const float*)d_in[3];
  const float* bq = (const float*)d_in[4];
  const float* Wk = (const float*)d_in[5];
  const float* bk = (const float*)d_in[6];
  const float* Wv = (const float*)d_in[7];
  const float* bv = (const float*)d_in[8];

  const size_t NX = (size_t)BS * EE;   // 4M elements
  const size_t NW = (size_t)EE * EE;   // 1M elements
  bf16_t* xqb = (bf16_t*)d_ws;
  bf16_t* xkb = xqb + NX;
  bf16_t* xvb = xkb + NX;
  bf16_t* wqb = xvb + NX;
  bf16_t* wkb = wqb + NW;
  bf16_t* wvb = wkb + NW;
  bf16_t* qb  = wvb + NW;
  bf16_t* kb  = qb + NX;
  bf16_t* vtb = kb + NX;

  // Stage 0: f32 -> bf16 (each element converted exactly once)
  const int nx4 = (int)(NX / 4), nw4 = (int)(NW / 4);
  cvt_bf16_kernel<<<(nx4 + 255) / 256, 256, 0, stream>>>(queries, xqb, nx4);
  cvt_bf16_kernel<<<(nx4 + 255) / 256, 256, 0, stream>>>(keys,    xkb, nx4);
  cvt_bf16_kernel<<<(nx4 + 255) / 256, 256, 0, stream>>>(values,  xvb, nx4);
  cvt_bf16_kernel<<<(nw4 + 255) / 256, 256, 0, stream>>>(Wq, wqb, nw4);
  cvt_bf16_kernel<<<(nw4 + 255) / 256, 256, 0, stream>>>(Wk, wkb, nw4);
  cvt_bf16_kernel<<<(nw4 + 255) / 256, 256, 0, stream>>>(Wv, wvb, nw4);

  // Stage 1: projections (32x64 tile per wave, 4 waves/block)
  dim3 gproj(BS / 32, EE / 256, 1);    // (128, 4)
  qkv_gemm_kernel<<<gproj, 128, 0, stream>>>(xqb, wqb, bq, qb, 0);
  qkv_gemm_kernel<<<gproj, 128, 0, stream>>>(xkb, wkb, bk, kb, 0);
  qkv_gemm_kernel<<<gproj, 128, 0, stream>>>(xvb, wvb, bv, vtb, 1);

  // Stage 2: attention (64 query rows per block)
  dim3 gattn(SS / 64, BB * HH, 1);     // (32, 32)
  attn_kernel<<<gattn, 128, 0, stream>>>(qb, kb, vtb, (float*)d_out);
}
